// SRNN_26087631356032
// MI455X (gfx1250) — matrix operand order repaired and checked by reading, not compiled, not run on hardware
//
#include <hip/hip_runtime.h>
#include <hip/hip_bf16.h>
#include <math.h>

typedef _Float16 half_t;
typedef __attribute__((ext_vector_type(16))) _Float16 v16h;
typedef __attribute__((ext_vector_type(8)))  _Float16 v8h;
typedef __attribute__((ext_vector_type(8)))  float    v8f;

#define T_STEPS 256
#define BATCH   64
#define XD      128
#define HD      1024
#define ZD      128
#define TBROWS  (T_STEPS * BATCH)   // 16384
#define S_ELEMS (TBROWS * ZD)       // 2097152 (== TBROWS*XD)
#define RED_BLOCKS 1024

// ---------------------------------------------------------------------------
// WMMA helpers (gfx1250, wave32)
// ---------------------------------------------------------------------------
__device__ __forceinline__ v8f wmma_f16(v16h a, v16h b, v8f c) {
  // D = A(16x32 f16) * B(32x16 f16) + C(16x16 f32)
  return __builtin_amdgcn_wmma_f32_16x16x32_f16(
      /*neg_a=*/false, a, /*neg_b=*/false, b,
      /*c_mod=*/(short)0, c, /*reuse_a=*/false, /*reuse_b=*/false);
}

// A fragment: row-major activations [M,K]; lane m = lane&15,
// K chunks at k0 + half*8 and k0 + 16 + half*8 (ISA 16-bit A layout).
__device__ __forceinline__ v16h load_a_frag(const half_t* __restrict__ base,
                                            int ld, int row0, int k0, int lane) {
  const int m  = row0 + (lane & 15);
  const int kk = k0 + ((lane >> 4) << 3);
  const half_t* p = base + (size_t)m * ld + kk;
  v8h lo = *(const v8h*)(p);
  v8h hi = *(const v8h*)(p + 16);
  v16h r;
#pragma unroll
  for (int i = 0; i < 8; ++i) { r[i] = lo[i]; r[i + 8] = hi[i]; }
  return r;
}

// B fragment: weights row-major [N,K] (computes X @ W^T); lane n = lane&15,
// K chunks at k0 + half*8 and k0 + 16 + half*8 (matches A layout).
__device__ __forceinline__ v16h load_b_frag(const half_t* __restrict__ base,
                                            int ld, int n0, int k0, int lane) {
  const int n  = n0 + (lane & 15);
  const int kk = k0 + ((lane >> 4) << 3);
  const half_t* p = base + (size_t)n * ld + kk;
  v8h lo = *(const v8h*)(p);
  v8h hi = *(const v8h*)(p + 16);
  v16h r;
#pragma unroll
  for (int i = 0; i < 8; ++i) { r[i] = lo[i]; r[i + 8] = hi[i]; }
  return r;
}

__device__ __forceinline__ float sigm_f(float x) { return 1.f / (1.f + expf(-x)); }
__device__ __forceinline__ float softplus_f(float x) {
  return x > 0.f ? x + log1pf(expf(-x)) : log1pf(expf(x));
}
__device__ __forceinline__ float apply_act(float v, int act) {
  if (act == 1) return v > 0.f ? v : 0.f;       // relu
  if (act == 2) return sigm_f(v);               // sigmoid
  if (act == 3) return softplus_f(v);           // softplus
  return v;
}

// ---------------------------------------------------------------------------
// Generic WMMA GEMM:  C[M,N] = act( A1[M,K1]*W1[N,K1]^T + A2[M,K2]*W2[N,K2]^T + bias )
// Block = 256 threads (8 waves), block tile 64x64.
// revX: read A rows time-reversed in blocks of BATCH (block tile M aligns with B=64).
// ---------------------------------------------------------------------------
__global__ __launch_bounds__(256) void gemm_wmma_kernel(
    const half_t* __restrict__ A1, int lda1, int rev1,
    const half_t* __restrict__ A2, int lda2, int rev2,
    const half_t* __restrict__ W1, int ldw1,
    const half_t* __restrict__ W2, int ldw2,
    int K1, int K2, int Mtot,
    const float* __restrict__ bias,
    float*  __restrict__ C32, int ldc32,
    half_t* __restrict__ C16, int ldc16,
    int act) {
  const int lane  = threadIdx.x & 31;
  const int wave  = threadIdx.x >> 5;
  const int m_sub = wave & 3;        // 4 x 16 rows
  const int n_grp = wave >> 2;       // 2 x 32 cols
  const int mb = blockIdx.y * 64;
  const int nb = blockIdx.x * 64;
  const int nrow0 = nb + n_grp * 32;
  const int arow1 = (rev1 ? (Mtot - 64 - mb) : mb) + m_sub * 16;
  const int arow2 = (rev2 ? (Mtot - 64 - mb) : mb) + m_sub * 16;

  v8f acc0 = {}; v8f acc1 = {};

  for (int k = 0; k < K1; k += 32) {
    v16h a  = load_a_frag(A1, lda1, arow1, k, lane);
    v16h b0 = load_b_frag(W1, ldw1, nrow0,      k, lane);
    v16h b1 = load_b_frag(W1, ldw1, nrow0 + 16, k, lane);
    acc0 = wmma_f16(a, b0, acc0);
    acc1 = wmma_f16(a, b1, acc1);
  }
  for (int k = 0; k < K2; k += 32) {
    v16h a  = load_a_frag(A2, lda2, arow2, k, lane);
    v16h b0 = load_b_frag(W2, ldw2, nrow0,      k, lane);
    v16h b1 = load_b_frag(W2, ldw2, nrow0 + 16, k, lane);
    acc0 = wmma_f16(a, b0, acc0);
    acc1 = wmma_f16(a, b1, acc1);
  }

#pragma unroll
  for (int i = 0; i < 2; ++i) {
    v8f acc = i ? acc1 : acc0;
    const int ncol = nrow0 + i * 16 + (lane & 15);
    const float bv = bias ? bias[ncol] : 0.f;
#pragma unroll
    for (int r = 0; r < 8; ++r) {
      const int m = mb + m_sub * 16 + ((lane >> 4) << 3) + r;
      float v = apply_act(acc[r] + bv, act);
      if (C32) C32[(size_t)m * ldc32 + ncol] = v;
      if (C16) C16[(size_t)m * ldc16 + ncol] = (half_t)v;
    }
  }
}

// ---------------------------------------------------------------------------
// One GRU recurrence step: gh = h_prev @ Whh^T (3 gates), fused gate math.
// gi (input gates) precomputed. Grid = HD/64 = 16 blocks, 256 threads.
// ---------------------------------------------------------------------------
__global__ __launch_bounds__(256) void gru_step_kernel(
    const half_t* __restrict__ gi_t,     // [BATCH, 3*HD], ld = 3*HD
    const half_t* __restrict__ Whh,      // [3*HD, HD] f16 row-major
    const float*  __restrict__ h_prev,   // [BATCH, HD] f32
    const half_t* __restrict__ h_prev16, // [BATCH, HD] f16
    float*  __restrict__ h_out,
    half_t* __restrict__ h_out16) {
  const int lane  = threadIdx.x & 31;
  const int wave  = threadIdx.x >> 5;
  const int m_sub = wave & 3;
  const int n_grp = wave >> 2;
  const int nb = blockIdx.x * 64;
  const int nrow0 = nb + n_grp * 32;

  v8f aR0 = {}, aR1 = {}, aZ0 = {}, aZ1 = {}, aN0 = {}, aN1 = {};

  for (int k = 0; k < HD; k += 32) {
    v16h a   = load_a_frag(h_prev16, HD, m_sub * 16, k, lane);
    v16h bR0 = load_b_frag(Whh, HD, 0 * HD + nrow0,      k, lane);
    v16h bR1 = load_b_frag(Whh, HD, 0 * HD + nrow0 + 16, k, lane);
    v16h bZ0 = load_b_frag(Whh, HD, 1 * HD + nrow0,      k, lane);
    v16h bZ1 = load_b_frag(Whh, HD, 1 * HD + nrow0 + 16, k, lane);
    v16h bN0 = load_b_frag(Whh, HD, 2 * HD + nrow0,      k, lane);
    v16h bN1 = load_b_frag(Whh, HD, 2 * HD + nrow0 + 16, k, lane);
    aR0 = wmma_f16(a, bR0, aR0);
    aR1 = wmma_f16(a, bR1, aR1);
    aZ0 = wmma_f16(a, bZ0, aZ0);
    aZ1 = wmma_f16(a, bZ1, aZ1);
    aN0 = wmma_f16(a, bN0, aN0);
    aN1 = wmma_f16(a, bN1, aN1);
  }

#pragma unroll
  for (int i = 0; i < 2; ++i) {
    v8f accR = i ? aR1 : aR0;
    v8f accZ = i ? aZ1 : aZ0;
    v8f accN = i ? aN1 : aN0;
    const int ncol = nrow0 + i * 16 + (lane & 15);
#pragma unroll
    for (int r = 0; r < 8; ++r) {
      const int m = m_sub * 16 + ((lane >> 4) << 3) + r;
      const size_t gbase = (size_t)m * (3 * HD) + ncol;
      float gir = (float)gi_t[gbase];
      float giz = (float)gi_t[gbase + HD];
      float gin = (float)gi_t[gbase + 2 * HD];
      float rg = sigm_f(gir + accR[r]);
      float zg = sigm_f(giz + accZ[r]);
      float ng = tanhf(gin + rg * accN[r]);
      float hp = h_prev[(size_t)m * HD + ncol];
      float hn = (1.f - zg) * ng + zg * hp;
      h_out  [(size_t)m * HD + ncol] = hn;
      h_out16[(size_t)m * HD + ncol] = (half_t)hn;
    }
  }
}

// ---------------------------------------------------------------------------
// Elementwise / utility kernels
// ---------------------------------------------------------------------------
__global__ void f32_to_f16_kernel(const float* __restrict__ s, half_t* __restrict__ d, int n) {
  int i = blockIdx.x * blockDim.x + threadIdx.x;
  if (i < n) d[i] = (half_t)s[i];
}
__global__ void fill_zero_f32_kernel(float* p, int n) {
  int i = blockIdx.x * blockDim.x + threadIdx.x;
  if (i < n) p[i] = 0.f;
}
__global__ void fill_zero_f16_kernel(half_t* p, int n) {
  int i = blockIdx.x * blockDim.x + threadIdx.x;
  if (i < n) p[i] = (half_t)0.f;
}
__global__ void compute_z_kernel(const float* __restrict__ eps,
                                 const float* __restrict__ em,
                                 const float* __restrict__ es,
                                 half_t* __restrict__ z, int n) {
  int i = blockIdx.x * blockDim.x + threadIdx.x;
  if (i < n) z[i] = (half_t)(eps[i] * es[i] + em[i]);
}

// Deterministic two-stage reduction for kld & nll.
__global__ __launch_bounds__(256) void reduce_stage1_kernel(
    const float* __restrict__ em, const float* __restrict__ es,
    const float* __restrict__ pm, const float* __restrict__ ps,
    const float* __restrict__ y,  const float* __restrict__ dm,
    float* __restrict__ partK, float* __restrict__ partN, int n) {
  __shared__ float sk[256], sn[256];
  float accK = 0.f, accN = 0.f;
  for (int i = blockIdx.x * 256 + threadIdx.x; i < n; i += gridDim.x * 256) {
    float psv = ps[i], esv = es[i];
    float d = em[i] - pm[i];
    accK += 2.f * logf(psv) - 2.f * logf(esv) + (esv * esv + d * d) / (psv * psv) - 1.f;
    float yv = y[i], dmv = dm[i];
    accN -= yv * logf(dmv) + (1.f - yv) * logf(1.f - dmv);
  }
  sk[threadIdx.x] = accK; sn[threadIdx.x] = accN;
  __syncthreads();
  for (int s = 128; s > 0; s >>= 1) {
    if (threadIdx.x < s) {
      sk[threadIdx.x] += sk[threadIdx.x + s];
      sn[threadIdx.x] += sn[threadIdx.x + s];
    }
    __syncthreads();
  }
  if (threadIdx.x == 0) { partK[blockIdx.x] = sk[0]; partN[blockIdx.x] = sn[0]; }
}
__global__ __launch_bounds__(256) void reduce_stage2_kernel(
    const float* __restrict__ partK, const float* __restrict__ partN,
    float* __restrict__ out, int nblk) {
  __shared__ float sk[256], sn[256];
  float accK = 0.f, accN = 0.f;
  for (int i = threadIdx.x; i < nblk; i += 256) { accK += partK[i]; accN += partN[i]; }
  sk[threadIdx.x] = accK; sn[threadIdx.x] = accN;
  __syncthreads();
  for (int s = 128; s > 0; s >>= 1) {
    if (threadIdx.x < s) {
      sk[threadIdx.x] += sk[threadIdx.x + s];
      sn[threadIdx.x] += sn[threadIdx.x + s];
    }
    __syncthreads();
  }
  if (threadIdx.x == 0) { out[0] = 0.5f * sk[0]; out[1] = sn[0]; }
}

// ---------------------------------------------------------------------------
// Host launch
// ---------------------------------------------------------------------------
extern "C" void kernel_launch(void* const* d_in, const int* in_sizes, int n_in,
                              void* d_out, int out_size, void* d_ws, size_t ws_size,
                              hipStream_t stream) {
  (void)in_sizes; (void)n_in; (void)out_size; (void)ws_size;

  const float* x      = (const float*)d_in[0];
  const float* y      = (const float*)d_in[1];
  const float* eps    = (const float*)d_in[2];
  const float* h_Wih  = (const float*)d_in[3];
  const float* h_Whh  = (const float*)d_in[4];
  const float* a_Wih  = (const float*)d_in[5];
  const float* a_Whh  = (const float*)d_in[6];
  const float* enc_W1 = (const float*)d_in[7];
  const float* enc_b1 = (const float*)d_in[8];
  const float* enc_W2 = (const float*)d_in[9];
  const float* enc_b2 = (const float*)d_in[10];
  const float* encm_W = (const float*)d_in[11];
  const float* encm_b = (const float*)d_in[12];
  const float* encs_W = (const float*)d_in[13];
  const float* encs_b = (const float*)d_in[14];
  const float* pr_W   = (const float*)d_in[15];
  const float* pr_b   = (const float*)d_in[16];
  const float* prm_W  = (const float*)d_in[17];
  const float* prm_b  = (const float*)d_in[18];
  const float* prs_W  = (const float*)d_in[19];
  const float* prs_b  = (const float*)d_in[20];
  const float* dec_W1 = (const float*)d_in[21];
  const float* dec_b1 = (const float*)d_in[22];
  const float* dec_W2 = (const float*)d_in[23];
  const float* dec_b2 = (const float*)d_in[24];
  const float* decm_W = (const float*)d_in[25];
  const float* decm_b = (const float*)d_in[26];
  const float* decs_W = (const float*)d_in[27];
  const float* decs_b = (const float*)d_in[28];

  float* out = (float*)d_out;
  float* out_enc_mean = out + 2;
  float* out_enc_std  = out + 2 + (size_t)S_ELEMS;
  float* out_dec_mean = out + 2 + (size_t)2 * S_ELEMS;
  float* out_dec_std  = out + 2 + (size_t)3 * S_ELEMS;

  // --- workspace carve-up (256B aligned) ---
  char* wsb = (char*)d_ws;
  size_t off = 0;
  auto alloc = [&](size_t bytes) -> char* {
    char* p = wsb + off;
    off = (off + bytes + 255) & ~(size_t)255;
    return p;
  };
  half_t* w_hWih = (half_t*)alloc((size_t)3 * HD * XD * 2);
  half_t* w_hWhh = (half_t*)alloc((size_t)3 * HD * HD * 2);
  half_t* w_aWih = (half_t*)alloc((size_t)3 * HD * (XD + HD) * 2);
  half_t* w_aWhh = (half_t*)alloc((size_t)3 * HD * HD * 2);
  half_t* w_eW1  = (half_t*)alloc((size_t)HD * HD * 2);
  half_t* w_eW2  = (half_t*)alloc((size_t)HD * HD * 2);
  half_t* w_emW  = (half_t*)alloc((size_t)ZD * HD * 2);
  half_t* w_esW  = (half_t*)alloc((size_t)ZD * HD * 2);
  half_t* w_pW   = (half_t*)alloc((size_t)HD * HD * 2);
  half_t* w_pmW  = (half_t*)alloc((size_t)ZD * HD * 2);
  half_t* w_psW  = (half_t*)alloc((size_t)ZD * HD * 2);
  half_t* w_dW1  = (half_t*)alloc((size_t)HD * (HD + ZD) * 2);
  half_t* w_dW2  = (half_t*)alloc((size_t)HD * HD * 2);
  half_t* w_dmW  = (half_t*)alloc((size_t)XD * HD * 2);
  half_t* w_dsW  = (half_t*)alloc((size_t)XD * HD * 2);
  half_t* x16    = (half_t*)alloc((size_t)TBROWS * XD * 2);
  half_t* y16    = (half_t*)alloc((size_t)TBROWS * XD * 2);
  half_t* gi     = (half_t*)alloc((size_t)TBROWS * 3 * HD * 2);  // reused for both GRUs
  float*  h32    = (float*) alloc((size_t)(T_STEPS + 1) * BATCH * HD * 4);
  half_t* h16    = (half_t*)alloc((size_t)(T_STEPS + 1) * BATCH * HD * 2);
  float*  a32    = (float*) alloc((size_t)(T_STEPS + 1) * BATCH * HD * 4);
  half_t* a16    = (half_t*)alloc((size_t)(T_STEPS + 1) * BATCH * HD * 2);
  half_t* hid1   = (half_t*)alloc((size_t)TBROWS * HD * 2);      // enc hidden1, then dec hidden1
  half_t* hid2   = (half_t*)alloc((size_t)TBROWS * HD * 2);      // enc hidden2, then dec hidden2
  half_t* pri16  = (half_t*)alloc((size_t)TBROWS * HD * 2);
  half_t* z16    = (half_t*)alloc((size_t)TBROWS * ZD * 2);
  float*  prm32  = (float*) alloc((size_t)S_ELEMS * 4);
  float*  prs32  = (float*) alloc((size_t)S_ELEMS * 4);
  float*  partK  = (float*) alloc(RED_BLOCKS * 4);
  float*  partN  = (float*) alloc(RED_BLOCKS * 4);

  half_t* h16_data = h16 + (size_t)BATCH * HD;  // slot 0 = zeros (initial state)
  half_t* a16_data = a16 + (size_t)BATCH * HD;

  auto conv = [&](const float* s, half_t* d, size_t n) {
    f32_to_f16_kernel<<<dim3((unsigned)((n + 255) / 256)), dim3(256), 0, stream>>>(s, d, (int)n);
  };

  // 1) convert weights + inputs to f16
  conv(h_Wih,  w_hWih, (size_t)3 * HD * XD);
  conv(h_Whh,  w_hWhh, (size_t)3 * HD * HD);
  conv(a_Wih,  w_aWih, (size_t)3 * HD * (XD + HD));
  conv(a_Whh,  w_aWhh, (size_t)3 * HD * HD);
  conv(enc_W1, w_eW1,  (size_t)HD * HD);
  conv(enc_W2, w_eW2,  (size_t)HD * HD);
  conv(encm_W, w_emW,  (size_t)ZD * HD);
  conv(encs_W, w_esW,  (size_t)ZD * HD);
  conv(pr_W,   w_pW,   (size_t)HD * HD);
  conv(prm_W,  w_pmW,  (size_t)ZD * HD);
  conv(prs_W,  w_psW,  (size_t)ZD * HD);
  conv(dec_W1, w_dW1,  (size_t)HD * (HD + ZD));
  conv(dec_W2, w_dW2,  (size_t)HD * HD);
  conv(decm_W, w_dmW,  (size_t)XD * HD);
  conv(decs_W, w_dsW,  (size_t)XD * HD);
  conv(x, x16, (size_t)TBROWS * XD);
  conv(y, y16, (size_t)TBROWS * XD);

  // 2) zero initial GRU states (slot 0)
  fill_zero_f32_kernel<<<dim3((BATCH * HD + 255) / 256), dim3(256), 0, stream>>>(h32, BATCH * HD);
  fill_zero_f16_kernel<<<dim3((BATCH * HD + 255) / 256), dim3(256), 0, stream>>>(h16, BATCH * HD);
  fill_zero_f32_kernel<<<dim3((BATCH * HD + 255) / 256), dim3(256), 0, stream>>>(a32, BATCH * HD);
  fill_zero_f16_kernel<<<dim3((BATCH * HD + 255) / 256), dim3(256), 0, stream>>>(a16, BATCH * HD);

  const dim3 blk(256);
  const dim3 grid_gi(3 * HD / 64, TBROWS / 64);   // (48, 256)
  const dim3 grid_hh(HD / 64, TBROWS / 64);       // (16, 256)
  const dim3 grid_zh(ZD / 64, TBROWS / 64);       // (2, 256)

  // 3) gi_h = x @ h_Wih^T   (big parallel GEMM, f16 out)
  gemm_wmma_kernel<<<grid_gi, blk, 0, stream>>>(
      x16, XD, 0, nullptr, 0, 0,
      w_hWih, XD, nullptr, 0,
      XD, 0, TBROWS, nullptr,
      nullptr, 0, gi, 3 * HD, 0);

  // 4) forward GRU over x (sequential critical path)
  for (int t = 0; t < T_STEPS; ++t) {
    gru_step_kernel<<<dim3(HD / 64), blk, 0, stream>>>(
        gi + (size_t)t * BATCH * 3 * HD, w_hWhh,
        h32 + (size_t)t * BATCH * HD, h16 + (size_t)t * BATCH * HD,
        h32 + (size_t)(t + 1) * BATCH * HD, h16 + (size_t)(t + 1) * BATCH * HD);
  }

  // 5) gi_a = concat(y_rev, h_rev) @ a_Wih^T  (rev folded into A reads)
  gemm_wmma_kernel<<<grid_gi, blk, 0, stream>>>(
      y16, XD, 1, h16_data, HD, 1,
      w_aWih, XD + HD, w_aWih + XD, XD + HD,
      XD, HD, TBROWS, nullptr,
      nullptr, 0, gi, 3 * HD, 0);

  // 6) backward GRU (operates in reversed time; output a32/a16 is a_rev sequence)
  for (int t = 0; t < T_STEPS; ++t) {
    gru_step_kernel<<<dim3(HD / 64), blk, 0, stream>>>(
        gi + (size_t)t * BATCH * 3 * HD, w_aWhh,
        a32 + (size_t)t * BATCH * HD, a16 + (size_t)t * BATCH * HD,
        a32 + (size_t)(t + 1) * BATCH * HD, a16 + (size_t)(t + 1) * BATCH * HD);
  }

  // 7) enc hidden1 = relu(a_fwd @ enc_W1^T + b1)   (a_fwd = reverse of a_rev)
  gemm_wmma_kernel<<<grid_hh, blk, 0, stream>>>(
      a16_data, HD, 1, nullptr, 0, 0,
      w_eW1, HD, nullptr, 0,
      HD, 0, TBROWS, enc_b1,
      nullptr, 0, hid1, HD, 1);
  // 8) enc = relu(hid1 @ enc_W2^T + b2)
  gemm_wmma_kernel<<<grid_hh, blk, 0, stream>>>(
      hid1, HD, 0, nullptr, 0, 0,
      w_eW2, HD, nullptr, 0,
      HD, 0, TBROWS, enc_b2,
      nullptr, 0, hid2, HD, 1);
  // 9) enc_mean / 10) enc_std
  gemm_wmma_kernel<<<grid_zh, blk, 0, stream>>>(
      hid2, HD, 0, nullptr, 0, 0,
      w_emW, HD, nullptr, 0,
      HD, 0, TBROWS, encm_b,
      out_enc_mean, ZD, nullptr, 0, 0);
  gemm_wmma_kernel<<<grid_zh, blk, 0, stream>>>(
      hid2, HD, 0, nullptr, 0, 0,
      w_esW, HD, nullptr, 0,
      HD, 0, TBROWS, encs_b,
      out_enc_std, ZD, nullptr, 0, 3);

  // 11) z = eps * enc_std + enc_mean  (f16 for dec GEMM)
  compute_z_kernel<<<dim3((S_ELEMS + 255) / 256), blk, 0, stream>>>(
      eps, out_enc_mean, out_enc_std, z16, S_ELEMS);

  // 12) pri = relu(h_rev @ pr_W^T + b)
  gemm_wmma_kernel<<<grid_hh, blk, 0, stream>>>(
      h16_data, HD, 1, nullptr, 0, 0,
      w_pW, HD, nullptr, 0,
      HD, 0, TBROWS, pr_b,
      nullptr, 0, pri16, HD, 1);
  // 13) prior_mean / 14) prior_std
  gemm_wmma_kernel<<<grid_zh, blk, 0, stream>>>(
      pri16, HD, 0, nullptr, 0, 0,
      w_pmW, HD, nullptr, 0,
      HD, 0, TBROWS, prm_b,
      prm32, ZD, nullptr, 0, 0);
  gemm_wmma_kernel<<<grid_zh, blk, 0, stream>>>(
      pri16, HD, 0, nullptr, 0, 0,
      w_psW, HD, nullptr, 0,
      HD, 0, TBROWS, prs_b,
      prs32, ZD, nullptr, 0, 3);

  // 15) dec hidden1 = relu(concat(z, h_rev) @ dec_W1^T + b1)
  gemm_wmma_kernel<<<grid_hh, blk, 0, stream>>>(
      z16, ZD, 0, h16_data, HD, 1,
      w_dW1, HD + ZD, w_dW1 + ZD, HD + ZD,
      ZD, HD, TBROWS, dec_b1,
      nullptr, 0, hid1, HD, 1);
  // 16) dec = relu(hid1 @ dec_W2^T + b2)
  gemm_wmma_kernel<<<grid_hh, blk, 0, stream>>>(
      hid1, HD, 0, nullptr, 0, 0,
      w_dW2, HD, nullptr, 0,
      HD, 0, TBROWS, dec_b2,
      nullptr, 0, hid2, HD, 1);
  // 17) dec_mean (sigmoid) / 18) dec_std (softplus)
  gemm_wmma_kernel<<<grid_zh, blk, 0, stream>>>(
      hid2, HD, 0, nullptr, 0, 0,
      w_dmW, HD, nullptr, 0,
      HD, 0, TBROWS, decm_b,
      out_dec_mean, XD, nullptr, 0, 2);
  gemm_wmma_kernel<<<grid_zh, blk, 0, stream>>>(
      hid2, HD, 0, nullptr, 0, 0,
      w_dsW, HD, nullptr, 0,
      HD, 0, TBROWS, decs_b,
      out_dec_std, XD, nullptr, 0, 3);

  // 19/20) deterministic kld + nll reductions
  reduce_stage1_kernel<<<dim3(RED_BLOCKS), blk, 0, stream>>>(
      out_enc_mean, out_enc_std, prm32, prs32, y, out_dec_mean,
      partK, partN, S_ELEMS);
  reduce_stage2_kernel<<<dim3(1), blk, 0, stream>>>(partK, partN, out, RED_BLOCKS);
}
